// GCN_89627377533178
// MI455X (gfx1250) — compile-verified, matrix operand order checked by reading, not run on hardware
//
#include <hip/hip_runtime.h>
#include <hip/hip_bf16.h>

// ---------------------------------------------------------------------------
// 3-layer GCN (GCNConv x3 + ReLU + global mean pool) for MI455X (gfx1250).
//
// Roofline: the edge scatter (~1.6 GB of L2 traffic per layer over the
// 51.2 MB h/agg arrays, 192 MB L2-resident) dominates; the GEMMs are only
// ~10 GFLOP total. GEMM uses v_wmma_f32_16x16x32_f16 with f32 accumulate on
// pre-converted f16 operands; W is pre-packed into B-fragment layout so the
// inner loop is pure b128 loads + WMMA. Scatter uses relaxed agent-scope
// float atomics (native global_atomic_add_f32).
// ---------------------------------------------------------------------------

#define DIMV 128
#define NGRAPH 512

typedef __attribute__((ext_vector_type(16))) _Float16 v16h;
typedef __attribute__((ext_vector_type(8)))  _Float16 v8h;
typedef __attribute__((ext_vector_type(8)))  float    v8f;

__device__ __forceinline__ void atomAddF(float* p, float v) {
  __hip_atomic_fetch_add(p, v, __ATOMIC_RELAXED, __HIP_MEMORY_SCOPE_AGENT);
}

// ---------------- elementwise helpers ----------------
__global__ __launch_bounds__(256) void k_fill(float* __restrict__ p, float v, int n) {
  int i = blockIdx.x * 256 + threadIdx.x;
  if (i < n) p[i] = v;
}

// 8 elements per thread; n must be a multiple of 8 (N*128 always is).
__global__ __launch_bounds__(256) void k_f32_to_f16_v8(const float* __restrict__ s,
                                                       _Float16* __restrict__ d, int n8) {
  int i = blockIdx.x * 256 + threadIdx.x;
  if (i >= n8) return;
  const float4 a = *(const float4*)(s + (size_t)i * 8);
  const float4 b = *(const float4*)(s + (size_t)i * 8 + 4);
  v8h o;
  o[0] = (_Float16)a.x; o[1] = (_Float16)a.y; o[2] = (_Float16)a.z; o[3] = (_Float16)a.w;
  o[4] = (_Float16)b.x; o[5] = (_Float16)b.y; o[6] = (_Float16)b.z; o[7] = (_Float16)b.w;
  *(v8h*)(d + (size_t)i * 8) = o;
}

// ---------------- degree / norm ----------------
__global__ __launch_bounds__(256) void k_degree(const int* __restrict__ dst,
                                                float* __restrict__ deg, int E) {
  int e = blockIdx.x * 256 + threadIdx.x;
  if (e < E) atomAddF(&deg[dst[e]], 1.0f);
}

__global__ __launch_bounds__(256) void k_rsqrt(const float* __restrict__ deg,
                                               float* __restrict__ dinv, int n) {
  int i = blockIdx.x * 256 + threadIdx.x;
  if (i < n) dinv[i] = rsqrtf(deg[i]);
}

__global__ __launch_bounds__(256) void k_edge_norm(const int* __restrict__ src,
                                                   const int* __restrict__ dst,
                                                   const float* __restrict__ dinv,
                                                   float* __restrict__ enorm, int E) {
  int e = blockIdx.x * 256 + threadIdx.x;
  if (e < E) enorm[e] = dinv[src[e]] * dinv[dst[e]];
}

// ---------------- W pre-pack into B-fragment layout ----------------
// Wp[colTile(8)][kstep(4)][lane(32)][16 halves]: lane's 16 B elements for one
// 16x16x32 WMMA, contiguous (two b128 loads in the GEMM).
// Element e (0..7):  B[k = kb+e     ][n = ct*16 + l15]
// Element e (8..15): B[k = kb+16+e-8][n = ct*16 + l15],  kb = 32*kk + 8*(lane>=16)
__global__ __launch_bounds__(256)
void k_pack_w(const float* __restrict__ W, _Float16* __restrict__ Wp) {
  int t = blockIdx.x * 256 + threadIdx.x;   // 0..1023
  if (t >= 1024) return;
  const int lane = t & 31;
  const int kk   = (t >> 5) & 3;
  const int ct   = t >> 7;
  const int l15  = lane & 15;
  const int hi   = lane >> 4;
  const int kb   = kk * 32 + hi * 8;
  const int col  = ct * 16 + l15;
  _Float16* o = Wp + (size_t)t * 16;
#pragma unroll
  for (int e = 0; e < 8; ++e) o[e]     = (_Float16)W[(kb + e) * DIMV + col];
#pragma unroll
  for (int e = 0; e < 8; ++e) o[8 + e] = (_Float16)W[(kb + 16 + e) * DIMV + col];
}

// ---------------- WMMA GEMM: H[N,128] = Xh[N,128] * W[128,128] ----------------
// Block = 256 threads = 8 waves; block covers a 16-row strip; wave w computes
// the 16x16 tile of columns [16w, 16w+16). K=128 in four 16x16x32 WMMA steps.
__global__ __launch_bounds__(256)
void k_gemm_wmma(const _Float16* __restrict__ Xh, const _Float16* __restrict__ Wp,
                 float* __restrict__ H) {
  const int lane = threadIdx.x & 31;
  const int wave = threadIdx.x >> 5;
  const int l15  = lane & 15;
  const int hi   = lane >> 4;            // 0: K+{0..7,16..23}; 1: K+{8..15,24..31}
  const int rowbase = blockIdx.x << 4;
  const int colbase = wave << 4;

  const _Float16* arow = Xh + (size_t)(rowbase + l15) * DIMV;
  v8f c = {};

#pragma unroll
  for (int kk = 0; kk < 4; ++kk) {
    const int kb = kk * 32 + hi * 8;
    // A fragment: two 16-byte contiguous runs of 8 halves per lane.
    v16h a;
    *(v8h*)&a       = *(const v8h*)(arow + kb);
    *((v8h*)&a + 1) = *(const v8h*)(arow + kb + 16);
    // B fragment: pre-packed, one contiguous 32-byte run per lane.
    const v16h b = *(const v16h*)(Wp + (size_t)(((wave * 4 + kk) * 32) + lane) * 16);

    c = __builtin_amdgcn_wmma_f32_16x16x32_f16(
        /*neg_a=*/false, a, /*neg_b=*/false, b,
        /*c_mod=*/(short)0, c, /*reuse_a=*/false, /*reuse_b=*/false);
  }

  // D layout: VGPR r -> row rowbase + r + 8*hi, col colbase + l15.
  float* outp = H + (size_t)(rowbase + hi * 8) * DIMV + colbase + l15;
#pragma unroll
  for (int r = 0; r < 8; ++r) outp[(size_t)r * DIMV] = c[r];
}

// ---------------- edge scatter: AGG[dst] += H[src] * enorm[e] ----------------
// One wave per edge; lane owns 4 dims (float4 gather + 4 float atomics).
__global__ __launch_bounds__(256)
void k_scatter(const float* __restrict__ H, const int* __restrict__ src,
               const int* __restrict__ dst, const float* __restrict__ enorm,
               float* __restrict__ AGG, int E) {
  int e = (blockIdx.x << 3) + (threadIdx.x >> 5);
  if (e >= E) return;
  const int lane = threadIdx.x & 31;
  const int s = src[e];
  const int d = dst[e];
  const float nrm = enorm[e];
  const float4 v = *(const float4*)(H + (size_t)s * DIMV + (lane << 2));
  float* ap = AGG + (size_t)d * DIMV + (lane << 2);
  atomAddF(ap + 0, v.x * nrm);
  atomAddF(ap + 1, v.y * nrm);
  atomAddF(ap + 2, v.z * nrm);
  atomAddF(ap + 3, v.w * nrm);
}

// ---------------- finalize: OUT = relu(AGG + H*dinv^2 + bias) ----------------
__global__ __launch_bounds__(256)
void k_finalize(const float* __restrict__ AGG, const float* __restrict__ H,
                const float* __restrict__ dinv, const float* __restrict__ bias,
                float* __restrict__ OUT, int n) {
  int i = blockIdx.x * 256 + threadIdx.x;
  if (i >= n) return;
  const int node = i >> 7;
  const int d    = i & (DIMV - 1);
  float sn = dinv[node];
  sn *= sn;
  float v = AGG[i] + H[i] * sn + bias[d];
  OUT[i] = fmaxf(v, 0.0f);
}

// ---------------- global mean pool ----------------
__global__ __launch_bounds__(256)
void k_pool_acc(const float* __restrict__ X, const int* __restrict__ batch,
                float* __restrict__ sums, float* __restrict__ counts, int n) {
  int i = blockIdx.x * 256 + threadIdx.x;
  if (i >= n) return;
  const int node = i >> 7;
  const int d    = i & (DIMV - 1);
  const int g    = batch[node];
  atomAddF(&sums[(size_t)g * DIMV + d], X[i]);
  if (d == 0) atomAddF(&counts[g], 1.0f);
}

__global__ __launch_bounds__(256)
void k_pool_div(const float* __restrict__ sums, const float* __restrict__ counts,
                float* __restrict__ out, int n) {
  int i = blockIdx.x * 256 + threadIdx.x;
  if (i >= n) return;
  const int g = i >> 7;
  out[i] = sums[i] / fmaxf(counts[g], 1.0f);
}

// ---------------------------------------------------------------------------
extern "C" void kernel_launch(void* const* d_in, const int* in_sizes, int n_in,
                              void* d_out, int out_size, void* d_ws, size_t ws_size,
                              hipStream_t stream) {
  const float* x     = (const float*)d_in[0];
  const int*   eidx  = (const int*)d_in[1];
  const int*   batch = (const int*)d_in[2];
  const float* W[3]  = {(const float*)d_in[3], (const float*)d_in[5], (const float*)d_in[7]};
  const float* B[3]  = {(const float*)d_in[4], (const float*)d_in[6], (const float*)d_in[8]};

  const int N = in_sizes[0] / DIMV;      // 100000
  const int E = in_sizes[1] / 2;         // 1600000
  const int* src = eidx;
  const int* dst = eidx + E;

  const size_t ND = (size_t)N * DIMV;    // 12.8M floats

  // ---- workspace layout (floats / halves); all blocks stay 32B-aligned ----
  float* ws   = (float*)d_ws;
  float* h    = ws;                                  // [N,128] f32
  float* agg  = h   + ND;                            // [N,128] f32
  float* xb   = agg + ND;                            // [N,128] f32
  float* deg  = xb  + ND;                            // [N]
  float* dinv = deg + N;                             // [N]
  float* sums = dinv + N;                            // [512,128]
  float* cnts = sums + (size_t)NGRAPH * DIMV;        // [512]
  float* enorm = cnts + NGRAPH;                      // [E]
  _Float16* Xh = (_Float16*)(enorm + ((E + 7) & ~7)); // [N,128] f16
  _Float16* Wp = Xh + ND;                            // [3][1024][16] f16 packed

  float* outNode  = (float*)d_out;                   // [N,128]
  float* outGraph = outNode + ND;                    // [512,128]

  const int TB = 256;
  const int gN   = (N + TB - 1) / TB;
  const int gE   = (E + TB - 1) / TB;
  const int gND  = (int)((ND + TB - 1) / TB);
  const int gND8 = (int)((ND / 8 + TB - 1) / TB);
  const int gGD  = (NGRAPH * DIMV + TB - 1) / TB;
  const int gGEMM = N / 16;                          // 6250 (N divisible by 16)
  const int gSCAT = (E + 7) / 8;                     // 8 edges (waves) per block

  // ---- degrees and symmetric norm ----
  k_fill<<<gN, TB, 0, stream>>>(deg, 1.0f, N);       // self-loop
  k_degree<<<gE, TB, 0, stream>>>(dst, deg, E);
  k_rsqrt<<<gN, TB, 0, stream>>>(deg, dinv, N);
  k_edge_norm<<<gE, TB, 0, stream>>>(src, dst, dinv, enorm, E);

  // ---- weights: pack into B-fragment layout (once) ----
  for (int l = 0; l < 3; ++l)
    k_pack_w<<<4, TB, 0, stream>>>(W[l], Wp + (size_t)l * DIMV * DIMV);

  // ---- 3 GCN layers; x sequence: x -> xb -> outNode -> outNode ----
  const float* xin[3]  = {x, xb, outNode};
  float*       xout[3] = {xb, outNode, outNode};
  for (int l = 0; l < 3; ++l) {
    k_f32_to_f16_v8<<<gND8, TB, 0, stream>>>(xin[l], Xh, (int)(ND / 8));
    k_gemm_wmma<<<gGEMM, TB, 0, stream>>>(Xh, Wp + (size_t)l * DIMV * DIMV, h);
    k_fill<<<gND, TB, 0, stream>>>(agg, 0.0f, (int)ND);
    k_scatter<<<gSCAT, TB, 0, stream>>>(h, src, dst, enorm, agg, E);
    k_finalize<<<gND, TB, 0, stream>>>(agg, h, dinv, B[l], xout[l], (int)ND);
  }

  // ---- global mean pool over batch ids ----
  k_fill<<<gGD, TB, 0, stream>>>(sums, 0.0f, NGRAPH * DIMV);
  k_fill<<<(NGRAPH + TB - 1) / TB, TB, 0, stream>>>(cnts, 0.0f, NGRAPH);
  k_pool_acc<<<gND, TB, 0, stream>>>(outNode, batch, sums, cnts, (int)ND);
  k_pool_div<<<gGD, TB, 0, stream>>>(sums, cnts, outGraph, NGRAPH * DIMV);
}
